// ClofNet_49357764165830
// MI455X (gfx1250) — matrix operand order repaired and checked by reading, not compiled, not run on hardware
//
#include <hip/hip_runtime.h>
#include <cstdint>
#include <cstddef>

// ---------------------------------------------------------------------------
// Types for CDNA5 WMMA (gfx1250, wave32)
// ---------------------------------------------------------------------------
typedef __bf16 v16bf __attribute__((ext_vector_type(16)));
typedef float  v8f   __attribute__((ext_vector_type(8)));
typedef unsigned short u16;
typedef unsigned int   u32;

#define DEV static __device__ __forceinline__

DEV u16 f2b(float f) {            // f32 -> bf16 bits, round-to-nearest-even
    union { float f; u32 u; } x; x.f = f;
    u32 r = (x.u + 0x7FFFu + ((x.u >> 16) & 1u)) >> 16;
    return (u16)r;
}
DEV float b2f(u16 b) {
    union { u32 u; float f; } x; x.u = ((u32)b) << 16;
    return x.f;
}
// order-preserving float <-> uint for atomicMax over signed floats
DEV u32 enc_ord(float v) { union { float f; u32 u; } x; x.f = v; return (x.u & 0x80000000u) ? ~x.u : (x.u | 0x80000000u); }
DEV float dec_ord(u32 u) { union { u32 u; float f; } x; x.u = (u & 0x80000000u) ? (u & 0x7FFFFFFFu) : ~u; return x.f; }

union Frag { v16bf v; uint4 u[2]; u16 s[16]; };

typedef __attribute__((address_space(3))) const u16 u16_as3;
DEV unsigned lds_off(const u16* p) { return (unsigned)(size_t)(u16_as3*)p; }

// Cooperative async copy of `nelem` u16 (multiple of 8) from global -> LDS.
// Uses CDNA5 GLOBAL_LOAD_ASYNC_TO_LDS_B128 (tracked by ASYNCcnt), then waits
// and barriers. Must be executed by all 256 threads of the block.
DEV void stage_to_lds_async(const u16* __restrict__ g, u16* l, int nelem) {
    for (int i = threadIdx.x * 8; i < nelem; i += 256 * 8) {
        unsigned lo = lds_off(l + i);
        const u16* ga = g + i;
        asm volatile("global_load_async_to_lds_b128 %0, %1, off"
                     :: "v"(lo), "v"(ga) : "memory");
    }
#if __has_builtin(__builtin_amdgcn_s_wait_asynccnt)
    __builtin_amdgcn_s_wait_asynccnt(0);
#else
    asm volatile("s_wait_asynccnt 0x0" ::: "memory");
#endif
    __syncthreads();
}

DEV v8f wmma_bf16(const Frag& a, const Frag& b, v8f c) {
    return __builtin_amdgcn_wmma_f32_16x16x32_bf16(false, a.v, false, b.v, (short)0, c, false, false);
}

// Shared epilogue: bias + activation (+resid), dual f32/bf16 stores, ldc = n_valid
template <int NT>
DEV void store_tile(const v8f (&acc)[NT], int m0, int lane, int hf,
                    const float* __restrict__ bias, float* __restrict__ Cf,
                    u16* __restrict__ Cb, const float* __restrict__ resid,
                    int n_valid, int act, int M)
{
    if (m0 >= M) return;
#pragma unroll
    for (int t = 0; t < NT; ++t) {
        int n = t * 16 + (lane & 15);
        if (n >= n_valid) continue;
        float bv = bias ? bias[n] : 0.0f;
#pragma unroll
        for (int r = 0; r < 8; ++r) {
            int mr = m0 + r + hf * 8;
            float v = acc[t][r] + bv;
            if (act == 1)      v = fmaxf(v, 0.0f);
            else if (act == 2) v = (v > 0.0f) ? v : (expf(v) - 1.0f);
            size_t idx = (size_t)mr * n_valid + n;
            if (resid) v += resid[idx];
            if (Cf) Cf[idx] = v;
            if (Cb) Cb[idx] = f2b(v);
        }
    }
}

// ---------------------------------------------------------------------------
// Generic bf16 WMMA GEMM:  C[M, n_valid] = act(A[M,Kpad] @ Wt^T + bias) (+resid)
//   A : bf16 row-major, lda = Kpad (multiple of 32, 16B-aligned rows)
//   Wt: bf16 [NT*16, Kpad] = W^T, staged into LDS once per block (async copy),
//       B fragments then read via ds_load_b128.
//   block = 256 threads = 8 waves; wave computes one 16-row x NT*16-col tile,
//   with next-K A fragment and next-t B fragment software-pipelined.
//   act: 0 = none, 1 = relu, 2 = elu
// ---------------------------------------------------------------------------
template <int NT>
__global__ void gemm_bf16_kernel(const u16* __restrict__ A, int Kpad,
                                 const u16* __restrict__ Wt,
                                 const float* __restrict__ bias,
                                 float* __restrict__ Cf, u16* __restrict__ Cb,
                                 const float* __restrict__ resid,
                                 int M, int n_valid, int act)
{
    extern __shared__ u16 lds[];
    stage_to_lds_async(Wt, lds, NT * 16 * Kpad);

    const int lane = threadIdx.x & 31;
    const int wave = threadIdx.x >> 5;
    const int hf   = (lane < 16) ? 0 : 1;
    const int m0   = (blockIdx.x * 8 + wave) * 16;
    if (m0 >= M) return;
    const int mrow = m0 + (lane & 15);

    v8f acc[NT] = {};
    const u16* arow = A + (size_t)mrow * Kpad + hf * 8;
    const u16* lrow = lds + (size_t)(lane & 15) * Kpad + hf * 8;

    Frag af;
    af.u[0] = *(const uint4*)(arow);
    af.u[1] = *(const uint4*)(arow + 16);
    for (int k0 = 0; k0 < Kpad; k0 += 32) {
        Frag afn = af;
        if (k0 + 32 < Kpad) {                   // prefetch next K-chunk of A
            afn.u[0] = *(const uint4*)(arow + k0 + 32);
            afn.u[1] = *(const uint4*)(arow + k0 + 48);
        }
        Frag bf;
        bf.u[0] = *(const uint4*)(lrow + k0);
        bf.u[1] = *(const uint4*)(lrow + k0 + 16);
#pragma unroll
        for (int t = 0; t < NT; ++t) {
            Frag bfn = bf;
            if (t + 1 < NT) {                   // prefetch next B fragment (LDS)
                const u16* lb2 = lrow + (size_t)(t + 1) * 16 * Kpad + k0;
                bfn.u[0] = *(const uint4*)(lb2);
                bfn.u[1] = *(const uint4*)(lb2 + 16);
            }
            acc[t] = wmma_bf16(af, bf, acc[t]);
            bf = bfn;
        }
        af = afn;
    }
    store_tile<NT>(acc, m0, lane, hf, bias, Cf, Cb, resid, n_valid, act, M);
}

// ---------------------------------------------------------------------------
// Fused gather-GEMM for the per-layer edge MLP first layer:
//   X1[e,:] = relu( concat(h[row[e]](128), h[col[e]](128), radial(1), ef(64),
//                   zeros -> Kpad=352) @ edge_W1 + b1 )        (Nout = 128)
// Weight tile (128x352 bf16 = 88KB) staged in LDS; one 16-row tile per wave.
// ---------------------------------------------------------------------------
DEV void load_edge_chunk(Frag& af, int chunk, int hf,
                         const u16* hr, const u16* hc, const u16* ep, float rad)
{
    const int k0 = chunk * 32;
    if (chunk < 4) {
        af.u[0] = *(const uint4*)(hr + k0);
        af.u[1] = *(const uint4*)(hr + k0 + 16);
    } else if (chunk < 8) {
        af.u[0] = *(const uint4*)(hc + (k0 - 128));
        af.u[1] = *(const uint4*)(hc + (k0 - 128) + 16);
    } else {   // K in [256,352): radial @256, ef @[257,321), zero pad >=321
#pragma unroll
        for (int i = 0; i < 16; ++i) {
            int K = k0 + ((i < 8) ? (hf * 8 + i) : (16 + hf * 8 + (i - 8)));
            u16 bits;
            if (K == 256)                 bits = f2b(rad);
            else if (K > 256 && K <= 320) bits = ep[K - 257];
            else                          bits = 0;
            af.s[i] = bits;
        }
    }
}

__global__ void edge_gemm1_kernel(const u16* __restrict__ hb,
                                  const u16* __restrict__ ef,
                                  const float* __restrict__ radial,
                                  const int* __restrict__ row,
                                  const int* __restrict__ col,
                                  const u16* __restrict__ Wt,   // [128, 352]
                                  const float* __restrict__ bias,
                                  u16* __restrict__ Cb,
                                  int E)
{
    extern __shared__ u16 lds[];
    stage_to_lds_async(Wt, lds, 128 * 352);

    const int lane = threadIdx.x & 31;
    const int wave = threadIdx.x >> 5;
    const int hf   = (lane < 16) ? 0 : 1;
    const int m0   = (blockIdx.x * 8 + wave) * 16;
    if (m0 >= E) return;
    const int e = m0 + (lane & 15);

    const u16* hr = hb + (size_t)row[e] * 128 + hf * 8;
    const u16* hc = hb + (size_t)col[e] * 128 + hf * 8;
    const u16* ep = ef + (size_t)e * 64;
    const float rad = radial[e];
    const u16* lrow = lds + (size_t)(lane & 15) * 352 + hf * 8;

    v8f acc[8] = {};
    Frag af;
    load_edge_chunk(af, 0, hf, hr, hc, ep, rad);
    for (int chunk = 0; chunk < 11; ++chunk) {
        const int k0 = chunk * 32;
        Frag afn = af;
        if (chunk + 1 < 11) load_edge_chunk(afn, chunk + 1, hf, hr, hc, ep, rad);
        Frag bf;
        bf.u[0] = *(const uint4*)(lrow + k0);
        bf.u[1] = *(const uint4*)(lrow + k0 + 16);
#pragma unroll
        for (int t = 0; t < 8; ++t) {
            Frag bfn = bf;
            if (t + 1 < 8) {
                const u16* lb2 = lrow + (size_t)(t + 1) * 16 * 352 + k0;
                bfn.u[0] = *(const uint4*)(lb2);
                bfn.u[1] = *(const uint4*)(lb2 + 16);
            }
            acc[t] = wmma_bf16(af, bf, acc[t]);
            bf = bfn;
        }
        af = afn;
    }
    store_tile<8>(acc, m0, lane, hf, bias, nullptr, Cb, nullptr, 128, 1, E);
}

// ---------------------------------------------------------------------------
// Scalar helper kernels
// ---------------------------------------------------------------------------
__global__ void zero_u32_kernel(u32* p, size_t n) {
    size_t i = (size_t)blockIdx.x * 256 + threadIdx.x;
    if (i < n) p[i] = 0;
}
__global__ void f32_to_b16_kernel(const float* __restrict__ s, u16* __restrict__ d, size_t n) {
    size_t i = (size_t)blockIdx.x * 256 + threadIdx.x;
    if (i < n) d[i] = f2b(s[i]);
}
// Wt[n*Kpad + k] = bf16(W[k*Nout + n]), zero-padded
__global__ void xpose_w_kernel(const float* __restrict__ W, u16* __restrict__ Wt,
                               int K, int Nout, int Kpad, int NoutPad) {
    int idx = blockIdx.x * 256 + threadIdx.x;
    if (idx >= NoutPad * Kpad) return;
    int n = idx / Kpad, k = idx % Kpad;
    Wt[idx] = (n < Nout && k < K) ? f2b(W[(size_t)k * Nout + n]) : (u16)0;
}
__global__ void centroid_accum_kernel(const float* __restrict__ x, float* __restrict__ coord,
                                      float* __restrict__ cent, const int* __restrict__ nn, int N) {
    int i = blockIdx.x * 256 + threadIdx.x;
    if (i >= N) return;
    int g = i / nn[0];
#pragma unroll
    for (int j = 0; j < 3; ++j) {
        float xm = (x[(size_t)i * 9 + j] + x[(size_t)i * 9 + 3 + j] + x[(size_t)i * 9 + 6 + j]) * (1.0f / 3.0f);
        coord[(size_t)i * 3 + j] = xm;
        atomicAdd(&cent[g * 3 + j], xm);
    }
}
__global__ void centroid_sub_kernel(float* __restrict__ coord, const float* __restrict__ cent,
                                    const int* __restrict__ nn, int N) {
    int i = blockIdx.x * 256 + threadIdx.x;
    if (i >= N) return;
    int g = i / nn[0];
    float inv = 1.0f / (float)nn[0];
#pragma unroll
    for (int j = 0; j < 3; ++j) coord[(size_t)i * 3 + j] -= cent[g * 3 + j] * inv;
}

DEV void localframe(const float* cr, const float* cc,
                    float* d, float* cx, float* vt, float* rad_out) {
    float diff[3] = { cr[0] - cc[0], cr[1] - cc[1], cr[2] - cc[2] };
    float rad = diff[0]*diff[0] + diff[1]*diff[1] + diff[2]*diff[2];
    float cr3[3] = { cr[1]*cc[2] - cr[2]*cc[1], cr[2]*cc[0] - cr[0]*cc[2], cr[0]*cc[1] - cr[1]*cc[0] };
    float dn = sqrtf(rad) + 1.0f;
    float cn = sqrtf(cr3[0]*cr3[0] + cr3[1]*cr3[1] + cr3[2]*cr3[2]) + 1.0f;
#pragma unroll
    for (int j = 0; j < 3; ++j) { d[j] = diff[j] / dn; cx[j] = cr3[j] / cn; }
    vt[0] = d[1]*cx[2] - d[2]*cx[1];
    vt[1] = d[2]*cx[0] - d[0]*cx[2];
    vt[2] = d[0]*cx[1] - d[1]*cx[0];
    *rad_out = rad;
}

__global__ void edge_feat_kernel(const float* __restrict__ coord,
                                 const int* __restrict__ row, const int* __restrict__ col,
                                 float* __restrict__ feat, u32* __restrict__ maxc, int E) {
    int e = blockIdx.x * 256 + threadIdx.x;
    if (e >= E) return;
    const float* cr = coord + (size_t)row[e] * 3;
    const float* cc = coord + (size_t)col[e] * 3;
    float d[3], cx[3], vt[3], rad;
    localframe(cr, cc, d, cx, vt, &rad);
    float ci[3], cj[3];
    const float* basis[3] = { d, cx, vt };
#pragma unroll
    for (int i = 0; i < 3; ++i) {
        ci[i] = basis[i][0]*cr[0] + basis[i][1]*cr[1] + basis[i][2]*cr[2];
        cj[i] = basis[i][0]*cc[0] + basis[i][1]*cc[1] + basis[i][2]*cc[2];
    }
    float ni = sqrtf(ci[0]*ci[0] + ci[1]*ci[1] + ci[2]*ci[2]) + 1e-5f;
    float nj = sqrtf(cj[0]*cj[0] + cj[1]*cj[1] + cj[2]*cj[2]) + 1e-5f;
    float cosv = (ci[0]*cj[0] + ci[1]*cj[1] + ci[2]*cj[2]) / ni / nj;
    float s2 = 1.0f - cosv * cosv;
    union { u32 u; float f; } qn; qn.u = 0x7FC00000u;
    float sinv = (s2 >= 0.0f) ? sqrtf(s2) : qn.f;
    float f[8] = { sinv, cosv, ci[0], ci[1], ci[2], cj[0], cj[1], cj[2] };
    float lm = -3.4e38f;
#pragma unroll
    for (int k = 0; k < 8; ++k) {
        feat[(size_t)e * 8 + k] = f[k];
        if (f[k] == f[k]) lm = fmaxf(lm, f[k]);
    }
    atomicMax(maxc, enc_ord(lm));
}

__global__ void fuse_assemble_kernel(const float* __restrict__ ea, const float* __restrict__ feat,
                                     const u32* __restrict__ maxc, u16* __restrict__ A, size_t total) {
    size_t idx = (size_t)blockIdx.x * 256 + threadIdx.x;
    if (idx >= total) return;
    size_t e = idx >> 5; int k = (int)(idx & 31);
    float v;
    if (k < 16) v = ea[e * 16 + k];
    else if (k < 24) {
        v = feat[e * 8 + (k - 16)];
        if (v != v) v = 10.0f * dec_ord(*maxc);
    } else v = 0.0f;
    A[idx] = f2b(v);
}

__global__ void deg_kernel(const int* __restrict__ row, float* __restrict__ deg, int E) {
    int e = blockIdx.x * 256 + threadIdx.x;
    if (e < E) atomicAdd(&deg[row[e]], 1.0f);
}

__global__ void edge_geom_kernel(const float* __restrict__ coord,
                                 const int* __restrict__ row, const int* __restrict__ col,
                                 float* __restrict__ frames, float* __restrict__ radial, int E) {
    int e = blockIdx.x * 256 + threadIdx.x;
    if (e >= E) return;
    float d[3], cx[3], vt[3], rad;
    localframe(coord + (size_t)row[e] * 3, coord + (size_t)col[e] * 3, d, cx, vt, &rad);
    float* fr = frames + (size_t)e * 9;
#pragma unroll
    for (int j = 0; j < 3; ++j) { fr[j] = d[j]; fr[3 + j] = cx[j]; fr[6 + j] = vt[j]; }
    radial[e] = rad;
}

// per-edge scatter: agg[row] += m ; cdelta[row] += clip(frame @ coff, +-100)
__global__ void edge_scatter_kernel(const u16* __restrict__ m, const float* __restrict__ coff,
                                    const float* __restrict__ frames, const int* __restrict__ row,
                                    float* __restrict__ agg, float* __restrict__ cdelta, int E) {
    int e = blockIdx.x;
    int tid = threadIdx.x;
    int r = row[e];
    atomicAdd(&agg[(size_t)r * 128 + tid], b2f(m[(size_t)e * 128 + tid]));
    if (tid < 3) {
        const float* fr = frames + (size_t)e * 9;
        float t = fr[tid] * coff[(size_t)e * 3 + 0]
                + fr[3 + tid] * coff[(size_t)e * 3 + 1]
                + fr[6 + tid] * coff[(size_t)e * 3 + 2];
        t = fminf(fmaxf(t, -100.0f), 100.0f);
        atomicAdd(&cdelta[(size_t)r * 3 + tid], t);
    }
}

__global__ void coord_update_kernel(float* __restrict__ coord, const float* __restrict__ cdelta,
                                    const float* __restrict__ deg, const float* __restrict__ svel,
                                    const float* __restrict__ vel, int N) {
    int idx = blockIdx.x * 256 + threadIdx.x;
    if (idx >= N * 3) return;
    int i = idx / 3;
    coord[idx] += cdelta[idx] / fmaxf(deg[i], 1.0f) + svel[i] * vel[idx];
}

__global__ void nodecat_kernel(const u16* __restrict__ hb, const float* __restrict__ agg,
                               u16* __restrict__ nc, size_t total) {
    size_t idx = (size_t)blockIdx.x * 256 + threadIdx.x;
    if (idx >= total) return;
    size_t i = idx >> 8; int k = (int)(idx & 255);
    nc[idx] = (k < 128) ? hb[i * 128 + k] : f2b(agg[i * 128 + (k - 128)]);
}

// ---------------------------------------------------------------------------
// Launcher
// ---------------------------------------------------------------------------
extern "C" void kernel_launch(void* const* d_in, const int* in_sizes, int n_in,
                              void* d_out, int out_size, void* d_ws, size_t ws_size,
                              hipStream_t stream)
{
    const int N = in_sizes[0] / 64;    // 20000
    const int E = in_sizes[3] / 16;    // 320000
    const int L = 4;

    const float* h0  = (const float*)d_in[0];
    const float* x   = (const float*)d_in[1];
    const float* vel = (const float*)d_in[2];
    const float* ea  = (const float*)d_in[3];
    const int* edges = (const int*)d_in[4];
    const int* row = edges;
    const int* col = edges + E;
    const int* nn  = (const int*)d_in[n_in - 1];   // n_nodes scalar (device)

    // params pytree leaves in JAX sorted-key order
    int p = 5;
    const float* dm_W1  = (const float*)d_in[p + 0];   // [128,64]
    const float* dm_W2  = (const float*)d_in[p + 1];   // [64,32]
    const float* dm_W3  = (const float*)d_in[p + 2];   // [32,5]
    const float* dm_b1  = (const float*)d_in[p + 3];
    const float* dm_b2  = (const float*)d_in[p + 4];
    const float* dm_b3  = (const float*)d_in[p + 5];
    const float* emb_W1 = (const float*)d_in[p + 6];   // [64,32]
    const float* emb_W2 = (const float*)d_in[p + 7];   // [32,128]
    const float* emb_b1 = (const float*)d_in[p + 8];
    const float* emb_b2 = (const float*)d_in[p + 9];
    const float* fuse_W1 = (const float*)d_in[p + 10]; // [24,64]
    const float* fuse_W2 = (const float*)d_in[p + 11]; // [64,64]
    const float* fuse_b1 = (const float*)d_in[p + 12];
    const float* fuse_b2 = (const float*)d_in[p + 13];
    const float* L_coord_W1 = (const float*)d_in[p + 14]; // [4,128,128]
    const float* L_coord_W2 = (const float*)d_in[p + 15]; // [4,128,3]
    const float* L_coord_b1 = (const float*)d_in[p + 16]; // [4,128]
    const float* L_edge_W1  = (const float*)d_in[p + 17]; // [4,321,128]
    const float* L_edge_W2  = (const float*)d_in[p + 18]; // [4,128,128]
    const float* L_edge_b1  = (const float*)d_in[p + 19];
    const float* L_edge_b2  = (const float*)d_in[p + 20];
    const float* L_node_W1  = (const float*)d_in[p + 21]; // [4,256,128]
    const float* L_node_W2  = (const float*)d_in[p + 22]; // [4,128,128]
    const float* L_node_b1  = (const float*)d_in[p + 23];
    const float* L_node_b2  = (const float*)d_in[p + 24];
    const float* L_vel_W1   = (const float*)d_in[p + 25]; // [4,128,128]
    const float* L_vel_W2   = (const float*)d_in[p + 26]; // [4,128,1]
    const float* L_vel_b1   = (const float*)d_in[p + 27];
    const float* L_vel_b2   = (const float*)d_in[p + 28]; // [4,1]

    // ---- workspace bump allocator ----
    char* base = (char*)d_ws; size_t off = 0;
    auto alloc = [&](size_t bytes) -> void* {
        off = (off + 255) & ~(size_t)255;
        void* q = base + off; off += bytes; return q;
    };
    float* h     = (float*)alloc((size_t)N * 128 * 4);
    u16*   hb    = (u16*)  alloc((size_t)N * 128 * 2);
    float* coord = (float*)alloc((size_t)N * 3 * 4);
    float* cent  = (float*)alloc(4096 * 3 * 4);
    u32*   maxc  = (u32*)  alloc(256);
    float* frames= (float*)alloc((size_t)E * 9 * 4);
    float* radial= (float*)alloc((size_t)E * 4);
    u16*   ef16  = (u16*)  alloc((size_t)E * 64 * 2);
    u16*   X1    = (u16*)  alloc((size_t)E * 128 * 2);
    u16*   m16   = (u16*)  alloc((size_t)E * 128 * 2);
    float* coff  = (float*)alloc((size_t)E * 3 * 4);
    float* agg   = (float*)alloc((size_t)N * 128 * 4);
    float* cdelta= (float*)alloc((size_t)N * 3 * 4);
    float* deg   = (float*)alloc((size_t)N * 4);
    float* svel  = (float*)alloc((size_t)N * 4);
    u16* nodecat = (u16*)  alloc((size_t)N * 256 * 2);
    u16* z1      = (u16*)  alloc((size_t)N * 64 * 2);
    u16* z2      = (u16*)  alloc((size_t)N * 32 * 2);
    // aliases into dead regions
    u16*   h0b     = X1;                       // [N,64] bf16, consumed by embedding
    float* feat    = (float*)X1;               // [E,8] f32, after embedding done
    u16*   fuseA   = m16;                      // [E,32] bf16
    u16*   fusehid = m16 + (size_t)E * 32;     // [E,64] bf16
    u16*   velhid  = nodecat;                  // [N,128] bf16 (before nodecat assembled)
    u16*   nodehid = m16;                      // [N,128] bf16 (edge m dead after scatter)

    // transposed bf16 weights
    u16* Wt_emb1 = (u16*)alloc(32 * 64 * 2);
    u16* Wt_emb2 = (u16*)alloc(128 * 32 * 2);
    u16* Wt_fu1  = (u16*)alloc(64 * 32 * 2);
    u16* Wt_fu2  = (u16*)alloc(64 * 64 * 2);
    u16* Wt_dm1  = (u16*)alloc(64 * 128 * 2);
    u16* Wt_dm2  = (u16*)alloc(32 * 64 * 2);
    u16* Wt_dm3  = (u16*)alloc(16 * 32 * 2);
    u16 *Wt_e1[4], *Wt_e2[4], *Wt_c1[4], *Wt_c2[4], *Wt_n1[4], *Wt_n2[4], *Wt_v1[4], *Wt_v2[4];
    for (int l = 0; l < L; ++l) {
        Wt_e1[l] = (u16*)alloc(128 * 352 * 2);
        Wt_e2[l] = (u16*)alloc(128 * 128 * 2);
        Wt_c1[l] = (u16*)alloc(128 * 128 * 2);
        Wt_c2[l] = (u16*)alloc(16 * 128 * 2);
        Wt_n1[l] = (u16*)alloc(128 * 256 * 2);
        Wt_n2[l] = (u16*)alloc(128 * 128 * 2);
        Wt_v1[l] = (u16*)alloc(128 * 128 * 2);
        Wt_v2[l] = (u16*)alloc(16 * 128 * 2);
    }

    auto zero = [&](void* q, size_t bytes) {
        size_t n = bytes / 4;
        zero_u32_kernel<<<(unsigned)((n + 255) / 256), 256, 0, stream>>>((u32*)q, n);
    };
    auto xpose = [&](const float* W, u16* Wt, int K, int Nout, int Kpad, int NoutPad) {
        int tot = NoutPad * Kpad;
        xpose_w_kernel<<<(tot + 255) / 256, 256, 0, stream>>>(W, Wt, K, Nout, Kpad, NoutPad);
    };
    auto gemm = [&](int NT, const u16* A, int Kpad, const u16* Wt, const float* bias,
                    float* Cf, u16* Cb, const float* resid, int M, int n_valid, int act) {
        dim3 g((M + 127) / 128);
        size_t smem = (size_t)NT * 16 * Kpad * 2;
        switch (NT) {
        case 1: gemm_bf16_kernel<1><<<g, 256, smem, stream>>>(A, Kpad, Wt, bias, Cf, Cb, resid, M, n_valid, act); break;
        case 2: gemm_bf16_kernel<2><<<g, 256, smem, stream>>>(A, Kpad, Wt, bias, Cf, Cb, resid, M, n_valid, act); break;
        case 4: gemm_bf16_kernel<4><<<g, 256, smem, stream>>>(A, Kpad, Wt, bias, Cf, Cb, resid, M, n_valid, act); break;
        default: gemm_bf16_kernel<8><<<g, 256, smem, stream>>>(A, Kpad, Wt, bias, Cf, Cb, resid, M, n_valid, act); break;
        }
    };

    // ---- weight prep (bf16, transposed, padded) ----
    xpose(emb_W1, Wt_emb1, 64, 32, 64, 32);
    xpose(emb_W2, Wt_emb2, 32, 128, 32, 128);
    xpose(fuse_W1, Wt_fu1, 24, 64, 32, 64);
    xpose(fuse_W2, Wt_fu2, 64, 64, 64, 64);
    xpose(dm_W1, Wt_dm1, 128, 64, 128, 64);
    xpose(dm_W2, Wt_dm2, 64, 32, 64, 32);
    xpose(dm_W3, Wt_dm3, 32, 5, 32, 16);
    for (int l = 0; l < L; ++l) {
        xpose(L_edge_W1 + (size_t)l * 321 * 128, Wt_e1[l], 321, 128, 352, 128);
        xpose(L_edge_W2 + (size_t)l * 128 * 128, Wt_e2[l], 128, 128, 128, 128);
        xpose(L_coord_W1 + (size_t)l * 128 * 128, Wt_c1[l], 128, 128, 128, 128);
        xpose(L_coord_W2 + (size_t)l * 128 * 3,   Wt_c2[l], 128, 3, 128, 16);
        xpose(L_node_W1 + (size_t)l * 256 * 128,  Wt_n1[l], 256, 128, 256, 128);
        xpose(L_node_W2 + (size_t)l * 128 * 128,  Wt_n2[l], 128, 128, 128, 128);
        xpose(L_vel_W1 + (size_t)l * 128 * 128,   Wt_v1[l], 128, 128, 128, 128);
        xpose(L_vel_W2 + (size_t)l * 128 * 1,     Wt_v2[l], 128, 1, 128, 16);
    }

    // ---- node embedding: h = relu(h0 @ W1 + b1) @ W2 + b2 ----
    {
        size_t n = (size_t)N * 64;
        f32_to_b16_kernel<<<(unsigned)((n + 255) / 256), 256, 0, stream>>>(h0, h0b, n);
    }
    gemm(2, h0b, 64, Wt_emb1, emb_b1, nullptr, z2, nullptr, N, 32, 1);      // [N,32] bf16
    gemm(8, z2, 32, Wt_emb2, emb_b2, h, hb, nullptr, N, 128, 0);            // h f32 + bf16

    // ---- coords: xm - per-graph centroid ----
    zero(cent, 4096 * 3 * 4);
    centroid_accum_kernel<<<(N + 255) / 256, 256, 0, stream>>>(x, coord, cent, nn, N);
    centroid_sub_kernel<<<(N + 255) / 256, 256, 0, stream>>>(coord, cent, nn, N);

    // ---- edge scalarization -> fused edge features ef ----
    zero(maxc, 4);
    edge_feat_kernel<<<(E + 255) / 256, 256, 0, stream>>>(coord, row, col, feat, maxc, E);
    {
        size_t tot = (size_t)E * 32;
        fuse_assemble_kernel<<<(unsigned)((tot + 255) / 256), 256, 0, stream>>>(ea, feat, maxc, fuseA, tot);
    }
    gemm(4, fuseA, 32, Wt_fu1, fuse_b1, nullptr, fusehid, nullptr, E, 64, 1);
    gemm(4, fusehid, 64, Wt_fu2, fuse_b2, nullptr, ef16, nullptr, E, 64, 1);

    // ---- degree ----
    zero(deg, (size_t)N * 4);
    deg_kernel<<<(E + 255) / 256, 256, 0, stream>>>(row, deg, E);

    // ---- message-passing layers ----
    for (int l = 0; l < L; ++l) {
        edge_geom_kernel<<<(E + 255) / 256, 256, 0, stream>>>(coord, row, col, frames, radial, E);
        edge_gemm1_kernel<<<(E + 127) / 128, 256, 128 * 352 * 2, stream>>>(
            hb, ef16, radial, row, col, Wt_e1[l], L_edge_b1 + l * 128, X1, E);
        gemm(8, X1, 128, Wt_e2[l], L_edge_b2 + l * 128, nullptr, m16, nullptr, E, 128, 1);   // m
        gemm(8, m16, 128, Wt_c1[l], L_coord_b1 + l * 128, nullptr, X1, nullptr, E, 128, 1);
        gemm(1, X1, 128, Wt_c2[l], nullptr, coff, nullptr, nullptr, E, 3, 0);                // coff
        zero(agg, (size_t)N * 128 * 4);
        zero(cdelta, (size_t)N * 3 * 4);
        edge_scatter_kernel<<<E, 128, 0, stream>>>(m16, coff, frames, row, agg, cdelta, E);
        gemm(8, hb, 128, Wt_v1[l], L_vel_b1 + l * 128, nullptr, velhid, nullptr, N, 128, 1);
        gemm(1, velhid, 128, Wt_v2[l], L_vel_b2 + l, svel, nullptr, nullptr, N, 1, 0);
        coord_update_kernel<<<(N * 3 + 255) / 256, 256, 0, stream>>>(coord, cdelta, deg, svel, vel, N);
        {
            size_t tot = (size_t)N * 256;
            nodecat_kernel<<<(unsigned)((tot + 255) / 256), 256, 0, stream>>>(hb, agg, nodecat, tot);
        }
        gemm(8, nodecat, 256, Wt_n1[l], L_node_b1 + l * 128, nullptr, nodehid, nullptr, N, 128, 1);
        gemm(8, nodehid, 128, Wt_n2[l], L_node_b2 + l * 128, h, hb, h, N, 128, 0);           // h += ...
    }

    // ---- decoder ----
    gemm(4, hb, 128, Wt_dm1, dm_b1, nullptr, z1, nullptr, N, 64, 2);   // elu
    gemm(2, z1, 64, Wt_dm2, dm_b2, nullptr, z2, nullptr, N, 32, 2);    // elu
    gemm(1, z2, 32, Wt_dm3, dm_b3, (float*)d_out, nullptr, nullptr, N, 5, 0);
    (void)ws_size; (void)out_size;
}
